// SelfAttention_13022340841873
// MI455X (gfx1250) — compile-verified, matrix operand order checked by reading, not run on hardware
//
#include <hip/hip_runtime.h>

// ---------------------------------------------------------------------------
// Shapes (fixed by the reference)
// ---------------------------------------------------------------------------
#define BB 2
#define SS 2048
#define DD 2048
#define HH 16
#define HD 128
#define HHD (HH * HD)   // 2048

typedef __attribute__((ext_vector_type(16))) __bf16 v16bf;
typedef __attribute__((ext_vector_type(8)))  float  v8f;
typedef __attribute__((ext_vector_type(2)))  __bf16 bf16x2;

// K-pair offset inside a 16-bit A/B fragment (ISA 7.12.2: 16-bit A 16x32):
// VGPR v (pair 2v,2v+1): K = {0,1},{2,3},{4,5},{6,7},{16,17},{18,19},{20,21},{22,23}
// lanes 16-31 add +8 to K.
__device__ __forceinline__ int kpair(int v, int hi) {
    return (v < 4 ? 2 * v : 16 + 2 * (v - 4)) + hi * 8;
}

// ---------------------------------------------------------------------------
// CDNA5 async global->LDS copy (bypasses VGPRs, tracked by ASYNCcnt).
// LDS destination address = low 32 bits of the generic pointer (flat->LDS
// truncation per ISA 10.2).  16-byte transfer per lane.
// ---------------------------------------------------------------------------
__device__ __forceinline__ void async_g2l_b128(void* lds_dst, const void* gsrc) {
    unsigned int ldsoff = (unsigned int)(unsigned long long)lds_dst;
    asm volatile("global_load_async_to_lds_b128 %0, %1, off"
                 :
                 : "v"(ldsoff), "v"(gsrc)
                 : "memory");
}
__device__ __forceinline__ void wait_async0() {
    asm volatile("s_wait_asynccnt 0" ::: "memory");
}

// ---------------------------------------------------------------------------
// Elementwise f32 -> bf16
// ---------------------------------------------------------------------------
__global__ void f32_to_bf16_kernel(const float* __restrict__ in,
                                   __bf16* __restrict__ out, int n) {
    int i = blockIdx.x * blockDim.x + threadIdx.x;
    if (i < n) out[i] = (__bf16)in[i];
}

// ---------------------------------------------------------------------------
// RoPE (rotate-half), applied in place to Q and K (bf16, [B*S, H*HD])
// one thread per (b, s, h, i<64) -> handles the (i, i+64) pair for Q and K
// ---------------------------------------------------------------------------
__global__ void rope_kernel(__bf16* __restrict__ Q, __bf16* __restrict__ K, int total) {
    int idx = blockIdx.x * blockDim.x + threadIdx.x;
    if (idx >= total) return;
    int i = idx & 63;
    int h = (idx >> 6) & (HH - 1);
    int s = (idx >> 10) & (SS - 1);
    int b = idx >> 21;
    // freq_i = 10000^(-2i/128) = exp(-i * ln(10000)/64)
    float ang = (float)s * __expf(-(float)i * 0.14391156516f);
    float cs, sn;
    __sincosf(ang, &sn, &cs);
    size_t base = ((size_t)(b * SS + s)) * HHD + h * HD;
    {
        float x1 = (float)Q[base + i], x2 = (float)Q[base + i + 64];
        Q[base + i]      = (__bf16)(x1 * cs - x2 * sn);
        Q[base + i + 64] = (__bf16)(x2 * cs + x1 * sn);
    }
    {
        float x1 = (float)K[base + i], x2 = (float)K[base + i + 64];
        K[base + i]      = (__bf16)(x1 * cs - x2 * sn);
        K[base + i + 64] = (__bf16)(x2 * cs + x1 * sn);
    }
}

// ---------------------------------------------------------------------------
// bf16 GEMM via v_wmma_f32_16x16x32_bf16.
// C[M,N] = A[M,K] * B[K,N].  Block tile 128x128, 8 waves: wave w owns rows
// [w*16, w*16+16) x all 128 cols (8 accum fragments).
// A tile staged via async global->LDS (layout-preserving), B tile staged
// through VGPRs because it is stored transposed ([n][k]) so the B-fragment
// K-pairs are contiguous b32/b128 LDS loads.
// ---------------------------------------------------------------------------
template <typename OutT>
__global__ __launch_bounds__(256) void gemm_bf16_wmma(
    const __bf16* __restrict__ A, const __bf16* __restrict__ Bm,
    OutT* __restrict__ C, int M, int N, int Kd) {
    __shared__ __align__(16) __bf16 Atile[128][32];
    __shared__ __align__(16) __bf16 Btile[128][32];  // [n][k]

    const int t = threadIdx.x;
    const int wave = t >> 5, lane = t & 31;
    const int l = lane & 15, hi = lane >> 4;
    const int m0 = blockIdx.y * 128, n0 = blockIdx.x * 128;

    // per-thread staging coordinates
    const int amr = t >> 1, akc = (t & 1) * 16;          // A: 128 rows x 32 k
    const int bkr = t >> 3, bnc = (t & 7) * 16;          // B: 32 k rows x 128 n

    v8f acc[8] = {};

    for (int k0 = 0; k0 < Kd; k0 += 32) {
        __syncthreads();
        {   // A tile: two async 16B global->LDS transfers per thread
            const __bf16* src = A + (size_t)(m0 + amr) * Kd + k0 + akc;
            async_g2l_b128(&Atile[amr][akc], src);
            async_g2l_b128(&Atile[amr][akc + 8], src + 8);
            if (k0 + 32 < Kd) __builtin_prefetch(src + 32, 0, 1);
        }
        {   // B tile: read 16 contiguous cols of one k-row, store transposed
            const __bf16* src = Bm + (size_t)(k0 + bkr) * N + n0 + bnc;
            #pragma unroll
            for (int j = 0; j < 16; ++j) Btile[bnc + j][bkr] = src[j];
            if (k0 + 32 < Kd) __builtin_prefetch(src + (size_t)32 * N, 0, 1);
        }
        wait_async0();
        __syncthreads();

        v16bf a;
        const int arow = wave * 16 + l;
        #pragma unroll
        for (int v = 0; v < 8; ++v) {
            bf16x2 p = *reinterpret_cast<const bf16x2*>(&Atile[arow][kpair(v, hi)]);
            a[2 * v] = p.x; a[2 * v + 1] = p.y;
        }
        #pragma unroll
        for (int sub = 0; sub < 8; ++sub) {
            v16bf bf;
            const int bn = sub * 16 + l;
            #pragma unroll
            for (int v = 0; v < 8; ++v) {
                bf16x2 p = *reinterpret_cast<const bf16x2*>(&Btile[bn][kpair(v, hi)]);
                bf[2 * v] = p.x; bf[2 * v + 1] = p.y;
            }
            acc[sub] = __builtin_amdgcn_wmma_f32_16x16x32_bf16(
                false, a, false, bf, (short)0, acc[sub], false, false);
        }
    }

    // C/D layout: VGPR r -> row r (lanes 0-15) / row r+8 (lanes 16-31), col = l
    #pragma unroll
    for (int sub = 0; sub < 8; ++sub)
        #pragma unroll
        for (int r = 0; r < 8; ++r) {
            int row = m0 + wave * 16 + r + hi * 8;
            int col = n0 + sub * 16 + l;
            C[(size_t)row * N + col] = (OutT)acc[sub][r];
        }
}

// ---------------------------------------------------------------------------
// Flash attention with WMMA.  Block = 8 waves, handles (b, h, 128 q-rows);
// wave w owns q-rows [q0+16w, q0+16w+16).  K tile staged via async
// global->LDS (layout-preserving); V tile staged transposed through VGPRs.
// Online softmax; P re-laid-out through per-wave LDS scratch into A-fragment
// layout.  All waves in a block share the same causal trip count.
// ---------------------------------------------------------------------------
__global__ __launch_bounds__(256) void attn_flash_wmma(
    const __bf16* __restrict__ Qb, const __bf16* __restrict__ Kb,
    const __bf16* __restrict__ Vb, __bf16* __restrict__ ctx) {
    __shared__ __align__(16) __bf16 Kt[32][128];       // [key][hd]
    __shared__ __align__(16) __bf16 Vt[128][32];       // [hd][key] (transposed)
    __shared__ __align__(16) __bf16 Plds[8][16][32];   // per-wave P scratch

    const int t = threadIdx.x;
    const int wave = t >> 5, lane = t & 31;
    const int l = lane & 15, hi = lane >> 4;
    const int b = blockIdx.z, h = blockIdx.y, q0 = blockIdx.x * 128;
    const int qw = q0 + wave * 16;
    const float scale = 0.08838834764831845f;  // 1/sqrt(128)

    // Q fragments for this wave's 16 rows: 4 x (16x32 bf16 A-fragment)
    v16bf qf[4];
    #pragma unroll
    for (int f = 0; f < 4; ++f)
        #pragma unroll
        for (int v = 0; v < 8; ++v) {
            bf16x2 p = *reinterpret_cast<const bf16x2*>(
                Qb + (size_t)(b * SS + qw + l) * HHD + h * HD + f * 32 + kpair(v, hi));
            qf[f][2 * v] = p.x; qf[f][2 * v + 1] = p.y;
        }

    v8f O[8] = {};
    float mrow[8], lrow[8];
    #pragma unroll
    for (int r = 0; r < 8; ++r) { mrow[r] = -3.0e38f; lrow[r] = 0.0f; }

    const int key_ = t >> 3, c0_ = (t & 7) * 16;  // staging coords (32 x 128)

    const int ntiles = (q0 + 128) >> 5;  // causal: keys < q0+128
    for (int kt = 0; kt < ntiles; ++kt) {
        const int k0 = kt * 32;
        __syncthreads();
        {   // cooperative K/V tile load (32 keys x 128 hd each)
            size_t g = (size_t)(b * SS + k0 + key_) * HHD + h * HD + c0_;
            // K: layout-preserving -> async direct to LDS
            async_g2l_b128(&Kt[key_][c0_], Kb + g);
            async_g2l_b128(&Kt[key_][c0_ + 8], Kb + g + 8);
            // V: transposed store through VGPRs
            const __bf16* vs = Vb + g;
            #pragma unroll
            for (int j = 0; j < 16; ++j) Vt[c0_ + j][key_] = vs[j];
            if (kt + 1 < ntiles) {
                __builtin_prefetch(Kb + g + (size_t)32 * HHD, 0, 1);
                __builtin_prefetch(Vb + g + (size_t)32 * HHD, 0, 1);
            }
        }
        wait_async0();
        __syncthreads();

        // scores: S(16x32) = Q(16x128) * K_tileT, two 16-key halves
        v8f s0 = {}, s1 = {};
        #pragma unroll
        for (int f = 0; f < 4; ++f) {
            v16bf b0, b1;
            #pragma unroll
            for (int v = 0; v < 8; ++v) {
                int kp = f * 32 + kpair(v, hi);
                bf16x2 p0 = *reinterpret_cast<const bf16x2*>(&Kt[l][kp]);
                bf16x2 p1 = *reinterpret_cast<const bf16x2*>(&Kt[16 + l][kp]);
                b0[2 * v] = p0.x; b0[2 * v + 1] = p0.y;
                b1[2 * v] = p1.x; b1[2 * v + 1] = p1.y;
            }
            s0 = __builtin_amdgcn_wmma_f32_16x16x32_bf16(false, qf[f], false, b0,
                                                         (short)0, s0, false, false);
            s1 = __builtin_amdgcn_wmma_f32_16x16x32_bf16(false, qf[f], false, b1,
                                                         (short)0, s1, false, false);
        }

        // online softmax update per row (row = r + 8*hi; its 16 cols live on
        // one 16-lane half-wave -> width-16 shuffle reductions)
        #pragma unroll
        for (int r = 0; r < 8; ++r) {
            int qidx = qw + r + hi * 8;
            float v0 = s0[r] * scale; if (k0 + l > qidx)      v0 = -3.0e38f;
            float v1 = s1[r] * scale; if (k0 + 16 + l > qidx) v1 = -3.0e38f;
            float mx = fmaxf(v0, v1);
            #pragma unroll
            for (int off = 8; off; off >>= 1) mx = fmaxf(mx, __shfl_xor(mx, off, 16));
            float mnew = fmaxf(mrow[r], mx);
            float p0 = __expf(v0 - mnew);
            float p1 = __expf(v1 - mnew);
            float alpha = __expf(mrow[r] - mnew);
            float rs = p0 + p1;
            #pragma unroll
            for (int off = 8; off; off >>= 1) rs += __shfl_xor(rs, off, 16);
            lrow[r] = lrow[r] * alpha + rs;
            mrow[r] = mnew;
            #pragma unroll
            for (int f2 = 0; f2 < 8; ++f2) O[f2][r] = O[f2][r] * alpha;
            Plds[wave][r + hi * 8][l]      = (__bf16)p0;
            Plds[wave][r + hi * 8][16 + l] = (__bf16)p1;
        }
        __syncthreads();

        // O(16x128) += P(16x32) * V(32x128)
        v16bf pf;
        #pragma unroll
        for (int v = 0; v < 8; ++v) {
            bf16x2 p = *reinterpret_cast<const bf16x2*>(&Plds[wave][l][kpair(v, hi)]);
            pf[2 * v] = p.x; pf[2 * v + 1] = p.y;
        }
        #pragma unroll
        for (int ch = 0; ch < 8; ++ch) {
            v16bf vf;
            #pragma unroll
            for (int v = 0; v < 8; ++v) {
                bf16x2 p = *reinterpret_cast<const bf16x2*>(&Vt[ch * 16 + l][kpair(v, hi)]);
                vf[2 * v] = p.x; vf[2 * v + 1] = p.y;
            }
            O[ch] = __builtin_amdgcn_wmma_f32_16x16x32_bf16(false, pf, false, vf,
                                                            (short)0, O[ch], false, false);
        }
    }

    // epilogue: ctx[b, q, h, hd] = O / l
    #pragma unroll
    for (int ch = 0; ch < 8; ++ch)
        #pragma unroll
        for (int r = 0; r < 8; ++r) {
            int row = qw + r + hi * 8;
            ctx[(size_t)(b * SS + row) * HHD + h * HD + ch * 16 + l] =
                (__bf16)(O[ch][r] / lrow[r]);
        }
}

// ---------------------------------------------------------------------------
// Host-side orchestration
// ---------------------------------------------------------------------------
extern "C" void kernel_launch(void* const* d_in, const int* in_sizes, int n_in,
                              void* d_out, int out_size, void* d_ws, size_t ws_size,
                              hipStream_t stream) {
    const float* x  = (const float*)d_in[0];
    // d_in[1] = mask (causal; implemented analytically)
    const float* Wq = (const float*)d_in[2];
    const float* Wk = (const float*)d_in[3];
    const float* Wv = (const float*)d_in[4];
    const float* Wo = (const float*)d_in[5];

    const size_t MB = 1ull << 20;
    char* ws = (char*)d_ws;
    __bf16* xb  = (__bf16*)(ws + 0 * MB);   // 16 MB; reused as ctx after QKV
    __bf16* Wqb = (__bf16*)(ws + 16 * MB);  // 8 MB each
    __bf16* Wkb = (__bf16*)(ws + 24 * MB);
    __bf16* Wvb = (__bf16*)(ws + 32 * MB);
    __bf16* Wob = (__bf16*)(ws + 40 * MB);
    __bf16* Qb  = (__bf16*)(ws + 48 * MB);  // 16 MB each
    __bf16* Kb  = (__bf16*)(ws + 64 * MB);
    __bf16* Vb  = (__bf16*)(ws + 80 * MB);
    __bf16* ctx = xb;

    const int nX = BB * SS * DD;    // 8388608
    const int nW = DD * HHD;        // 4194304

    f32_to_bf16_kernel<<<(nX + 255) / 256, 256, 0, stream>>>(x,  xb,  nX);
    f32_to_bf16_kernel<<<(nW + 255) / 256, 256, 0, stream>>>(Wq, Wqb, nW);
    f32_to_bf16_kernel<<<(nW + 255) / 256, 256, 0, stream>>>(Wk, Wkb, nW);
    f32_to_bf16_kernel<<<(nW + 255) / 256, 256, 0, stream>>>(Wv, Wvb, nW);
    f32_to_bf16_kernel<<<(nW + 255) / 256, 256, 0, stream>>>(Wo, Wob, nW);

    const int M = BB * SS;          // 4096
    dim3 ggrid(HHD / 128, M / 128); // (16, 32)
    gemm_bf16_wmma<__bf16><<<ggrid, 256, 0, stream>>>(xb, Wqb, Qb, M, HHD, DD);
    gemm_bf16_wmma<__bf16><<<ggrid, 256, 0, stream>>>(xb, Wkb, Kb, M, HHD, DD);
    gemm_bf16_wmma<__bf16><<<ggrid, 256, 0, stream>>>(xb, Wvb, Vb, M, HHD, DD);

    const int nR = BB * SS * HH * 64;  // 4194304
    rope_kernel<<<(nR + 255) / 256, 256, 0, stream>>>(Qb, Kb, nR);

    dim3 agrid(SS / 128, HH, BB);      // (16, 16, 2)
    attn_flash_wmma<<<agrid, 256, 0, stream>>>(Qb, Kb, Vb, ctx);

    gemm_bf16_wmma<float><<<ggrid, 256, 0, stream>>>(ctx, Wob, (float*)d_out, M, DD, HHD);
}
